// Conv2D_77687368450730
// MI455X (gfx1250) — 8-bit matrix operand order repaired and checked by reading, compiled, not run on hardware
//
#include <hip/hip_runtime.h>

typedef __attribute__((ext_vector_type(8))) int v8i;

#define TILE_H 32
#define TILE_W 32
#define INTILE 34
#define CIN    32
#define H_IN   226
#define W_IN   226
#define H_OUT  224
#define W_OUT  224
#define KSIZE  288
#define KCH    5

__global__ __launch_bounds__(256) void conv2d_q8_wmma(
    const int* __restrict__ in, const int* __restrict__ filt,
    const int* __restrict__ bias, const int* __restrict__ qm_p,
    const int* __restrict__ exp_p, const int* __restrict__ zp_p,
    float* __restrict__ out)
{
    __shared__ unsigned char s_in[INTILE * INTILE * CIN];
    __shared__ unsigned int  s_fb[KCH * 16];

    const int tid  = threadIdx.x;
    const int lane = tid & 31;
    const int wave = tid >> 5;
    const int half = lane >> 4;
    const int mrow = lane & 15;

    const int blk = blockIdx.x;
    const int tc  = blk % (W_OUT / TILE_W);
    const int tr  = (blk / (W_OUT / TILE_W)) % (H_OUT / TILE_H);
    const int b   = blk / ((W_OUT / TILE_W) * (H_OUT / TILE_H));
    const int r0  = tr * TILE_H, c0 = tc * TILE_W;

    const int* in_b = in + (size_t)b * H_IN * W_IN * CIN;
    for (int u = tid; u < INTILE * INTILE * (CIN / 4); u += 256) {
        int row = u / (INTILE * (CIN / 4));
        int rem = u - row * (INTILE * (CIN / 4));
        int col = rem >> 3;
        int c4  = rem & 7;
        const int4 iv = *(const int4*)(in_b + ((size_t)(r0 + row) * W_IN + (c0 + col)) * CIN + c4 * 4);
        unsigned int packed = (unsigned)(iv.x & 255)
                            | ((unsigned)(iv.y & 255) << 8)
                            | ((unsigned)(iv.z & 255) << 16)
                            | ((unsigned)(iv.w & 255) << 24);
        *(unsigned int*)&s_in[(row * INTILE + col) * CIN + c4 * 4] = packed;
    }

    if (tid < KCH * 16) {
        int kc = tid >> 4;
        int d  = tid & 15;
        int hh = d >> 3;
        int p  = d & 7;
        unsigned int packed = 0;
        #pragma unroll
        for (int j = 0; j < 4; ++j) {
            int k   = kc * 64 + ((p >> 1) * 16) + hh * 8 + (p & 1) * 4 + j;
            int val = (k < KSIZE) ? filt[b * KSIZE + k] : 0;
            packed |= ((unsigned int)(val & 255)) << (8 * j);
        }
        s_fb[tid] = packed;
    }
    __syncthreads();

    const int qm   = qm_p[0];
    const int expo = exp_p[0];
    const int zp   = zp_p[0];
    const int bb   = bias[b];
    const int red_m = (qm < 2147418112) ? ((qm + (1 << 15)) >> 16) : 32767;
    const int ts    = 15 - expo;
    const long long rnd = 1ll << (ts - 1);

    v8i bmat[KCH];
    #pragma unroll
    for (int kc = 0; kc < KCH; ++kc) {
        #pragma unroll
        for (int p = 0; p < 8; ++p)
            bmat[kc][p] = (int)s_fb[kc * 16 + half * 8 + p];
    }

    const bool writer = ((mrow >> 3) == half);

    for (int g = 0; g < 4; ++g) {
        const int rowrel = wave * 4 + g;
        for (int ch = 0; ch < 2; ++ch) {
            const int colbase = ch * 16;
            v8i acc = {bb, bb, bb, bb, bb, bb, bb, bb};
            #pragma unroll
            for (int kc = 0; kc < KCH; ++kc) {
                v8i a;
                #pragma unroll
                for (int p = 0; p < 8; ++p) {
                    int k0 = kc * 64 + ((p >> 1) * 16) + half * 8 + (p & 1) * 4;
                    unsigned int aw = 0;
                    if (k0 < KSIZE) {
                        int dy = k0 / 96;
                        int t2 = k0 - dy * 96;
                        int dx = t2 >> 5;
                        int cs = t2 & 31;
                        aw = *(const unsigned int*)&s_in[((rowrel + dy) * INTILE
                                                          + (colbase + mrow + dx)) * CIN + cs];
                    }
                    a[p] = (int)aw;
                }
                acc = __builtin_amdgcn_wmma_i32_16x16x64_iu8(true, a, true, bmat[kc], acc, false, false);
            }

            const int s2 = mrow & 7;
            int e01 = (s2 & 1) ? acc[1] : acc[0];
            int e23 = (s2 & 1) ? acc[3] : acc[2];
            int e45 = (s2 & 1) ? acc[5] : acc[4];
            int e67 = (s2 & 1) ? acc[7] : acc[6];
            int e03 = (s2 & 2) ? e23 : e01;
            int e47 = (s2 & 2) ? e67 : e45;
            int sel = (s2 & 4) ? e47 : e03;

            if (writer) {
                long long x = (long long)sel * red_m + rnd;
                int xi = (int)(x >> ts) + zp;
                xi = (xi < -128) ? -128 : xi;
                xi = (xi > 127) ? 127 : xi;
                const int r = r0 + rowrel;
                out[((size_t)b * H_OUT + r) * W_OUT + (c0 + colbase + mrow)] = (float)xi;
            }
        }
    }
}

extern "C" void kernel_launch(void* const* d_in, const int* in_sizes, int n_in,
                              void* d_out, int out_size, void* d_ws, size_t ws_size,
                              hipStream_t stream) {
    const int* in   = (const int*)d_in[0];
    const int* filt = (const int*)d_in[1];
    const int* bias = (const int*)d_in[2];
    const int* qm   = (const int*)d_in[3];
    const int* ex   = (const int*)d_in[4];
    const int* zp   = (const int*)d_in[5];
    float* out = (float*)d_out;

    dim3 grid(64 * (H_OUT / TILE_H) * (W_OUT / TILE_W));
    conv2d_q8_wmma<<<grid, 256, 0, stream>>>(in, filt, bias, qm, ex, zp, out);
}
